// CLUB_51144470560984
// MI455X (gfx1250) — matrix operand order repaired and checked by reading, not compiled, not run on hardware
//
#include <hip/hip_runtime.h>

typedef __attribute__((ext_vector_type(16))) _Float16 v16h;
typedef __attribute__((ext_vector_type(8)))  float    v8f;

#define EMB 128
#define HID 128
#define NPTS 1024

__device__ __forceinline__ float gelu_exact(float x) {
    return 0.5f * x * (1.0f + erff(x * 0.70710678118654752f));
}

// ---------------------------------------------------------------------------
// Prep 1: a[j][h] = dot(z1[j,:], W1[h, 0:128]);  b[i][h] = dot(z2[i,:], W1[h,128:256])
// ---------------------------------------------------------------------------
__global__ void club_prep_ab(const float* __restrict__ z1,
                             const float* __restrict__ z2,
                             const float* __restrict__ W1,
                             float* __restrict__ Abuf,
                             float* __restrict__ Bbuf) {
    int idx   = blockIdx.x * blockDim.x + threadIdx.x;   // 0 .. 2*N*HID-1
    int which = idx >> 17;                               // N*HID == 1<<17
    int r     = (idx >> 7) & (NPTS - 1);
    int h     = idx & (HID - 1);
    const float* zrow = (which ? z2 : z1) + r * EMB;
    const float* wrow = W1 + h * (2 * EMB) + (which ? EMB : 0);
    float s = 0.f;
#pragma unroll 8
    for (int d = 0; d < EMB; ++d) s += zrow[d] * wrow[d];
    (which ? Bbuf : Abuf)[r * HID + h] = s;
}

// ---------------------------------------------------------------------------
// Prep 2: pack W2^T (B[h,g] = W2[g,h]) into WMMA f16 B-fragment layout.
// Tile (t,n): K-range 32t..32t+31, N-range 16n..16n+15.
// Lane L: col = L&15; hi = L>>4; element e holds K=32t+8*hi+(e<8 ? e : 8+e).
// flat: w2f[((t*8+n)*32 + L)*16 + e]
// ---------------------------------------------------------------------------
__global__ void club_prep_w2(const float* __restrict__ W2,
                             _Float16* __restrict__ w2f) {
    int tl   = blockIdx.x * blockDim.x + threadIdx.x;    // 0..1023
    int tile = tl >> 5;                                  // (t*8 + n)
    int lane = tl & 31;
    int t    = tile >> 3;
    int n    = tile & 7;
    int hi   = (lane >> 4) & 1;
    int col  = lane & 15;
    int g    = 16 * n + col;
    int hb   = 32 * t + 8 * hi;
    _Float16* dst = w2f + (tile * 32 + lane) * 16;
#pragma unroll
    for (int e = 0; e < 16; ++e)
        dst[e] = (_Float16)W2[g * HID + hb + ((e < 8) ? e : (8 + e))];
}

// ---------------------------------------------------------------------------
// Main fused kernel: 8 waves/block, each wave = 16 output pairs (one i, 16 j).
// ---------------------------------------------------------------------------
__global__ void __launch_bounds__(256)
club_main(const float* __restrict__ Abuf,
          const float* __restrict__ Bbuf,
          const _Float16* __restrict__ w2f,
          const float* __restrict__ b1,
          const float* __restrict__ b2,
          const float* __restrict__ W3,
          const float* __restrict__ b3,
          float* __restrict__ out) {
    __shared__ __align__(32) _Float16 w2lds[32 * 32 * 16];   // 32 KB

    // Stage all W2 fragments into LDS (2048 uint4, 256 threads x 8)
    {
        const uint4* src = (const uint4*)w2f;
        uint4*       dst = (uint4*)w2lds;
#pragma unroll
        for (int k = 0; k < 8; ++k)
            dst[threadIdx.x + 256 * k] = src[threadIdx.x + 256 * k];
    }
    __syncthreads();

    const int wave = threadIdx.x >> 5;
    const int lane = threadIdx.x & 31;
    const int gwid = blockIdx.x * 8 + wave;
    const int i    = gwid >> 6;           // 0..1023
    const int j0   = (gwid & 63) << 4;    // 0..1008 step 16

    const int row = lane & 15;            // A-matrix row M
    const int hi  = (lane >> 4) & 1;      // lane-high half selects K offset

    const float* arow = Abuf + (j0 + row) * HID;
    const float* brow = Bbuf + i * HID;

    // Build A fragments: h1 = gelu(a[j] + b[i] + b1) in f16, ISA 16x32 layout.
    // lane, element e -> h = 32t + 8*hi + (e<8 ? e : 8+e)   [second run at +16]
    v16h afrag[4];
#pragma unroll
    for (int t = 0; t < 4; ++t) {
        const int hb = 32 * t + 8 * hi;
#pragma unroll
        for (int e = 0; e < 16; ++e) {
            const int h = hb + ((e < 8) ? e : (8 + e));   // e>=8: +16 gap
            const float x = arow[h] + brow[h] + b1[h];
            afrag[t][e] = (_Float16)gelu_exact(x);
        }
    }

    v8f acc[8];
#pragma unroll
    for (int n = 0; n < 8; ++n)
#pragma unroll
        for (int k = 0; k < 8; ++k) acc[n][k] = 0.f;

    // H2_pre(16x128) = H1(16x128) * W2^T : 4 K-tiles x 8 N-tiles of WMMA
#pragma unroll
    for (int t = 0; t < 4; ++t) {
#pragma unroll
        for (int n = 0; n < 8; ++n) {
            const v16h* bp =
                (const v16h*)(w2lds + (((t * 8 + n) * 32) + lane) * 16);
            acc[n] = __builtin_amdgcn_wmma_f32_16x16x32_f16(
                false, afrag[t], false, *bp, (short)0, acc[n], false, false);
        }
    }

    // Epilogue: gelu(acc + b2) dotted with W3, reduced over the 128 g's.
    // C layout: acc[n][r] = value at M = r + 8*hi, N(g) = 16n + (lane&15).
    const int col = lane & 15;
    float s[8];
#pragma unroll
    for (int r = 0; r < 8; ++r) s[r] = 0.f;
#pragma unroll
    for (int n = 0; n < 8; ++n) {
        const int g   = 16 * n + col;
        const float w3g = W3[g];
        const float b2g = b2[g];
#pragma unroll
        for (int r = 0; r < 8; ++r)
            s[r] += w3g * gelu_exact(acc[n][r] + b2g);
    }

    // Reduce across the 16 lanes of each half-wave (xor 1,2,4,8 stays in-half)
#pragma unroll
    for (int m = 1; m < 16; m <<= 1) {
#pragma unroll
        for (int r = 0; r < 8; ++r)
            s[r] += __shfl_xor(s[r], m, 32);
    }

    if ((lane & 15) == 0) {
        const float bias = b3[0];
        const int jb = j0 + 8 * hi;       // rows 0-7 from lane 0, 8-15 from lane 16
#pragma unroll
        for (int r = 0; r < 8; ++r)
            out[i * NPTS + jb + r] = s[r] + bias;
    }
}

// ---------------------------------------------------------------------------
extern "C" void kernel_launch(void* const* d_in, const int* in_sizes, int n_in,
                              void* d_out, int out_size, void* d_ws, size_t ws_size,
                              hipStream_t stream) {
    const float* z1 = (const float*)d_in[0];
    const float* z2 = (const float*)d_in[1];
    const float* W1 = (const float*)d_in[2];
    const float* b1 = (const float*)d_in[3];
    const float* W2 = (const float*)d_in[4];
    const float* b2 = (const float*)d_in[5];
    const float* W3 = (const float*)d_in[6];
    const float* b3 = (const float*)d_in[7];
    float* out = (float*)d_out;

    float*    Abuf = (float*)d_ws;                       // 1024*128 f32
    float*    Bbuf = Abuf + NPTS * HID;                  // 1024*128 f32
    _Float16* w2f  = (_Float16*)(Bbuf + NPTS * HID);     // 32*32*16 f16 = 32 KB

    club_prep_ab<<<(2 * NPTS * HID) / 256, 256, 0, stream>>>(z1, z2, W1, Abuf, Bbuf);
    club_prep_w2<<<4, 256, 0, stream>>>(W2, w2f);

    const int waveTiles = NPTS * (NPTS / 16);            // 65536
    club_main<<<waveTiles / 8, 256, 0, stream>>>(Abuf, Bbuf, w2f,
                                                 b1, b2, W3, b3, out);
}
